// AttentionBlock_19207093748382
// MI455X (gfx1250) — compile-verified, matrix operand order checked by reading, not run on hardware
//
#include <hip/hip_runtime.h>
#include <hip/hip_bf16.h>

// ---------------------------------------------------------------------------
// AttentionBlock for MI455X (gfx1250): GroupNorm -> QKV GEMM -> flash attn
// -> proj GEMM + residual.  All matmuls use v_wmma_f32_16x16x32_f16; GEMM
// A-tiles are staged to LDS with gfx1250 async global->LDS copies (ASYNCcnt),
// double-buffered against WMMA.
// ---------------------------------------------------------------------------

typedef __attribute__((ext_vector_type(16))) _Float16 v16h;
typedef __attribute__((ext_vector_type(8)))  float    v8f;
typedef int vi4 __attribute__((vector_size(16)));  // builtin's pointee type

constexpr int Bc  = 16;    // batch
constexpr int C   = 512;   // channels
constexpr int HW  = 1024;  // spatial (32*32)
constexpr int NH  = 8;     // heads
constexpr int HD  = 64;    // head dim
constexpr int G   = 8;     // groups
constexpr int CPG = 64;    // channels per group

#if defined(__has_builtin)
#if __has_builtin(__builtin_amdgcn_global_load_async_to_lds_b128)
#define HAVE_ASYNC_LDS 1
#endif
#endif

// Copy 32B (one fragment row-chunk pair) global -> LDS.
// Async path: global_load_async_to_lds_b128 x2 (same ptrs, imm offset 16 adds
// to both global and LDS address per ISA 10.x).  Fallback: VGPR round trip.
__device__ __forceinline__ void stage16x2(const _Float16* gsrc, _Float16* ldst) {
#ifdef HAVE_ASYNC_LDS
  __attribute__((address_space(1))) vi4* g =
      (__attribute__((address_space(1))) vi4*)(uintptr_t)gsrc;
  __attribute__((address_space(3))) vi4* l =
      (__attribute__((address_space(3))) vi4*)(uint32_t)(uintptr_t)ldst;
  __builtin_amdgcn_global_load_async_to_lds_b128(g, l, 0, 0);
  __builtin_amdgcn_global_load_async_to_lds_b128(g, l, 16, 0);
#else
  const uint4* s = reinterpret_cast<const uint4*>(gsrc);
  uint4* d = reinterpret_cast<uint4*>(ldst);
  d[0] = s[0];
  d[1] = s[1];
#endif
}

__device__ __forceinline__ void wait_stage() {
#ifdef HAVE_ASYNC_LDS
#if __has_builtin(__builtin_amdgcn_s_wait_asynccnt)
  __builtin_amdgcn_s_wait_asynccnt(0);
#else
  asm volatile("s_wait_asynccnt 0x0" ::: "memory");
#endif
#endif
}

// Load a 16x32-f16 A/B WMMA fragment: two contiguous 16B chunks per lane.
// Lane layout (ISA 7.12.2): lanes 0-15 row/col = lane, K 0-7 & 16-23;
// lanes 16-31 same rows, K 8-15 & 24-31 -> chunk offsets lhalf*8 and 16+lhalf*8.
__device__ __forceinline__ v16h load_frag2(const _Float16* p0, const _Float16* p1) {
  union { v16h v; uint4 u[2]; } r;
  r.u[0] = *reinterpret_cast<const uint4*>(p0);
  r.u[1] = *reinterpret_cast<const uint4*>(p1);
  return r.v;
}

__device__ __forceinline__ v8f wmma16(v16h a, v16h b, v8f c) {
  return __builtin_amdgcn_wmma_f32_16x16x32_f16(false, a, false, b, (short)0, c,
                                                false, false);
}

// ------------------------- weight fp32 -> fp16 -----------------------------
__global__ void k_cvt_w(const float* __restrict__ wq, const float* __restrict__ wp,
                        _Float16* __restrict__ wqh, _Float16* __restrict__ wph,
                        int nq, int np) {
  int i = blockIdx.x * blockDim.x + threadIdx.x;
  if (i < nq) wqh[i] = (_Float16)wq[i];
  if (i < np) wph[i] = (_Float16)wp[i];
}

// ------------------------- GroupNorm -> hT[b][n][c] f16 --------------------
// Pass 1: coalesced stats.  Pass 2: 64x64 LDS transpose tile so both the
// strided-x read and the transposed hT write are coalesced.
__global__ __launch_bounds__(256)
void k_gnorm(const float* __restrict__ x, const float* __restrict__ gw,
             const float* __restrict__ gb, _Float16* __restrict__ hT) {
  const int b = blockIdx.x / G, g = blockIdx.x % G;
  const float* xp = x + ((size_t)b * C + (size_t)g * CPG) * HW;
  const int tid = threadIdx.x;

  float s = 0.f, s2 = 0.f;
  for (int i = tid; i < CPG * HW; i += 256) {
    float v = xp[i];
    s += v; s2 += v * v;
  }
  __shared__ float rs[256], rq[256];
  __shared__ _Float16 tile[64 * 72];
  rs[tid] = s; rq[tid] = s2;
  __syncthreads();
  for (int st = 128; st > 0; st >>= 1) {
    if (tid < st) { rs[tid] += rs[tid + st]; rq[tid] += rq[tid + st]; }
    __syncthreads();
  }
  const float inv_n = 1.0f / (float)(CPG * HW);
  const float mean = rs[0] * inv_n;
  const float rstd = rsqrtf(rq[0] * inv_n - mean * mean + 1e-5f);

  const int c0 = g * CPG;
  for (int n0 = 0; n0 < HW; n0 += 64) {
    for (int idx = tid; idx < 64 * 64; idx += 256) {
      int cl = idx >> 6, nn = idx & 63;  // reads coalesced along nn
      float v = xp[(size_t)cl * HW + n0 + nn];
      tile[nn * 72 + cl] =
          (_Float16)((v - mean) * rstd * gw[c0 + cl] + gb[c0 + cl]);
    }
    __syncthreads();
    for (int idx = tid; idx < 64 * 64; idx += 256) {
      int nn = idx >> 6, cl = idx & 63;  // writes coalesced along cl
      hT[((size_t)b * HW + n0 + nn) * C + c0 + cl] = tile[nn * 72 + cl];
    }
    __syncthreads();
  }
}

// ------------------------- QKV GEMM (1536 x 1024 x 512 per batch) ----------
// A = qkv_w f16 (async double-buffered LDS), B = hT rows (direct frag loads).
// Epilogue scatters to qT/kT [b,h][n][d] (scale folded into q) and v[b,h][d][n].
__global__ __launch_bounds__(256)
void k_qkv(const _Float16* __restrict__ wqh, const float* __restrict__ qkv_b,
           const _Float16* __restrict__ hT,
           _Float16* __restrict__ qT, _Float16* __restrict__ kT,
           _Float16* __restrict__ vbuf) {
  __shared__ _Float16 ldsA[2][128 * 40];  // double-buffered 128x32 K-slab
  const int b = blockIdx.z;
  const int mBase = blockIdx.y * 128;
  const int nBase = blockIdx.x * 128;
  const int tid = threadIdx.x;
  const int w = tid >> 5, lane = tid & 31;
  const int wm = w >> 2, wn = w & 3;
  const int lhalf = lane >> 4, l15 = lane & 15;
  const int row = tid >> 1, seg = (tid & 1) * 16;
  const _Float16* arow = wqh + (size_t)(mBase + row) * C + seg;

  v8f acc[4][2] = {};

  stage16x2(arow, &ldsA[0][row * 40 + seg]);
  wait_stage();
  __syncthreads();

  for (int kk = 0; kk < C; kk += 32) {
    const int cur = (kk >> 5) & 1;
    if (kk + 32 < C)  // prefetch next K-slab into the other buffer
      stage16x2(arow + kk + 32, &ldsA[cur ^ 1][row * 40 + seg]);

    v16h bfr[2];
    for (int nt = 0; nt < 2; ++nt) {
      int j = nBase + wn * 32 + nt * 16 + l15;
      const _Float16* bp = hT + ((size_t)b * HW + j) * C + kk + lhalf * 8;
      bfr[nt] = load_frag2(bp, bp + 16);
      if (kk + 32 < C) __builtin_prefetch(bp + 32, 0, 1);
    }
    for (int mt = 0; mt < 4; ++mt) {
      const _Float16* ap =
          &ldsA[cur][(wm * 64 + mt * 16 + l15) * 40 + lhalf * 8];
      v16h afr = load_frag2(ap, ap + 16);
      for (int nt = 0; nt < 2; ++nt)
        acc[mt][nt] = wmma16(afr, bfr[nt], acc[mt][nt]);
    }
    wait_stage();
    __syncthreads();
  }

  for (int mt = 0; mt < 4; ++mt)
    for (int nt = 0; nt < 2; ++nt) {
      int n = nBase + wn * 32 + nt * 16 + l15;
      for (int v = 0; v < 8; ++v) {
        int o = mBase + wm * 64 + mt * 16 + v + lhalf * 8;  // D-layout row
        float val = acc[mt][nt][v] + qkv_b[o];
        int which = o >> 9, c = o & 511, head = c >> 6, d = c & 63;
        if (which == 0)
          qT[(((size_t)b * NH + head) * HW + n) * HD + d] =
              (_Float16)(val * 0.125f);
        else if (which == 1)
          kT[(((size_t)b * NH + head) * HW + n) * HD + d] = (_Float16)val;
        else
          vbuf[(((size_t)b * NH + head) * HD + d) * HW + n] = (_Float16)val;
      }
    }
}

// ------------------------- flash attention ---------------------------------
// One wave per 32-query tile; stream 32-key chunks; online softmax; P restaged
// f16 through wave-private LDS; O accumulated in f32.  Out: attnT[b][n][c].
__global__ __launch_bounds__(256)
void k_attn(const _Float16* __restrict__ qT, const _Float16* __restrict__ kT,
            const _Float16* __restrict__ vbuf, _Float16* __restrict__ attnT) {
  __shared__ _Float16 ldsP[8 * 32 * 40];  // 8 waves x (32x32 P tile, stride 40)
  const int w = threadIdx.x >> 5, lane = threadIdx.x & 31;
  const int wg = blockIdx.x * 8 + w;
  const int iBlk = wg & 31, head = (wg >> 5) & 7, b = wg >> 8;
  const int i0 = iBlk * 32;
  const int lhalf = lane >> 4, l15 = lane & 15;

  const _Float16* qbase = qT + (((size_t)b * NH + head) * HW) * HD;
  const _Float16* kbase = kT + (((size_t)b * NH + head) * HW) * HD;
  const _Float16* vbase = vbuf + (((size_t)b * NH + head) * HD) * HW;
  _Float16* ldsw = &ldsP[w * 32 * 40];

  v16h qf[2][2];
  for (int mi = 0; mi < 2; ++mi)
    for (int ks = 0; ks < 2; ++ks) {
      const _Float16* p =
          qbase + (size_t)(i0 + mi * 16 + l15) * HD + ks * 32 + lhalf * 8;
      qf[mi][ks] = load_frag2(p, p + 16);
    }

  v8f oacc[2][4] = {};
  float mrun[2][8], lrun[2][8];
  for (int mi = 0; mi < 2; ++mi)
    for (int v = 0; v < 8; ++v) { mrun[mi][v] = -1e30f; lrun[mi][v] = 0.f; }

  for (int j0 = 0; j0 < HW; j0 += 32) {
    v16h kf[2][2];
    for (int nj = 0; nj < 2; ++nj)
      for (int ks = 0; ks < 2; ++ks) {
        const _Float16* p =
            kbase + (size_t)(j0 + nj * 16 + l15) * HD + ks * 32 + lhalf * 8;
        kf[nj][ks] = load_frag2(p, p + 16);
      }

    v8f s[2][2] = {};
    for (int mi = 0; mi < 2; ++mi)
      for (int nj = 0; nj < 2; ++nj) {
        s[mi][nj] = wmma16(qf[mi][0], kf[nj][0], s[mi][nj]);
        s[mi][nj] = wmma16(qf[mi][1], kf[nj][1], s[mi][nj]);
      }

    for (int mi = 0; mi < 2; ++mi) {
      float corr[8];
      for (int v = 0; v < 8; ++v) {
        float t = fmaxf(s[mi][0][v], s[mi][1][v]);
        for (int off = 1; off < 16; off <<= 1)
          t = fmaxf(t, __shfl_xor(t, off, 32));
        float mn = fmaxf(mrun[mi][v], t);
        float cf = __expf(mrun[mi][v] - mn);
        float p0 = __expf(s[mi][0][v] - mn);
        float p1 = __expf(s[mi][1][v] - mn);
        s[mi][0][v] = p0;
        s[mi][1][v] = p1;
        float r = p0 + p1;
        for (int off = 1; off < 16; off <<= 1) r += __shfl_xor(r, off, 32);
        lrun[mi][v] = lrun[mi][v] * cf + r;
        mrun[mi][v] = mn;
        corr[v] = cf;
      }
      for (int nd = 0; nd < 4; ++nd)
        for (int v = 0; v < 8; ++v) oacc[mi][nd][v] *= corr[v];
      // D-layout -> row-major P tile in LDS (f16)
      for (int nj = 0; nj < 2; ++nj)
        for (int v = 0; v < 8; ++v) {
          int il = mi * 16 + v + lhalf * 8;
          int jl = nj * 16 + l15;
          ldsw[il * 40 + jl] = (_Float16)s[mi][nj][v];
        }
    }

    v16h pa[2];
    for (int mi = 0; mi < 2; ++mi) {
      const _Float16* p = &ldsw[(mi * 16 + l15) * 40 + lhalf * 8];
      pa[mi] = load_frag2(p, p + 16);
    }
    v16h vb[4];
    for (int nd = 0; nd < 4; ++nd) {
      const _Float16* p = vbase + (size_t)(nd * 16 + l15) * HW + j0 + lhalf * 8;
      vb[nd] = load_frag2(p, p + 16);
    }
    for (int mi = 0; mi < 2; ++mi)
      for (int nd = 0; nd < 4; ++nd)
        oacc[mi][nd] = wmma16(pa[mi], vb[nd], oacc[mi][nd]);
  }

  for (int mi = 0; mi < 2; ++mi)
    for (int v = 0; v < 8; ++v) {
      float inv = 1.0f / lrun[mi][v];
      int i = i0 + mi * 16 + v + lhalf * 8;
      for (int nd = 0; nd < 4; ++nd) {
        int c = head * 64 + nd * 16 + l15;
        attnT[((size_t)b * HW + i) * C + c] = (_Float16)(oacc[mi][nd][v] * inv);
      }
    }
}

// ------------------------- proj GEMM + bias + residual ---------------------
__global__ __launch_bounds__(256)
void k_proj(const _Float16* __restrict__ wph, const float* __restrict__ proj_b,
            const _Float16* __restrict__ aT, const float* __restrict__ x,
            float* __restrict__ out) {
  __shared__ _Float16 ldsA[2][128 * 40];
  const int b = blockIdx.z;
  const int mBase = blockIdx.y * 128;
  const int nBase = blockIdx.x * 128;
  const int tid = threadIdx.x;
  const int w = tid >> 5, lane = tid & 31;
  const int wm = w >> 2, wn = w & 3;
  const int lhalf = lane >> 4, l15 = lane & 15;
  const int row = tid >> 1, seg = (tid & 1) * 16;
  const _Float16* arow = wph + (size_t)(mBase + row) * C + seg;

  v8f acc[4][2] = {};

  stage16x2(arow, &ldsA[0][row * 40 + seg]);
  wait_stage();
  __syncthreads();

  for (int kk = 0; kk < C; kk += 32) {
    const int cur = (kk >> 5) & 1;
    if (kk + 32 < C)
      stage16x2(arow + kk + 32, &ldsA[cur ^ 1][row * 40 + seg]);

    v16h bfr[2];
    for (int nt = 0; nt < 2; ++nt) {
      int j = nBase + wn * 32 + nt * 16 + l15;
      const _Float16* bp = aT + ((size_t)b * HW + j) * C + kk + lhalf * 8;
      bfr[nt] = load_frag2(bp, bp + 16);
      if (kk + 32 < C) __builtin_prefetch(bp + 32, 0, 1);
    }
    for (int mt = 0; mt < 4; ++mt) {
      const _Float16* ap =
          &ldsA[cur][(wm * 64 + mt * 16 + l15) * 40 + lhalf * 8];
      v16h afr = load_frag2(ap, ap + 16);
      for (int nt = 0; nt < 2; ++nt)
        acc[mt][nt] = wmma16(afr, bfr[nt], acc[mt][nt]);
    }
    wait_stage();
    __syncthreads();
  }

  for (int mt = 0; mt < 4; ++mt)
    for (int nt = 0; nt < 2; ++nt) {
      int n = nBase + wn * 32 + nt * 16 + l15;
      for (int v = 0; v < 8; ++v) {
        int o = mBase + wm * 64 + mt * 16 + v + lhalf * 8;
        size_t idx = ((size_t)b * C + o) * HW + n;
        out[idx] = x[idx] + acc[mt][nt][v] + proj_b[o];
      }
    }
}

// ---------------------------------------------------------------------------
extern "C" void kernel_launch(void* const* d_in, const int* in_sizes, int n_in,
                              void* d_out, int out_size, void* d_ws, size_t ws_size,
                              hipStream_t stream) {
  (void)in_sizes; (void)n_in; (void)out_size; (void)ws_size;
  const float* x      = (const float*)d_in[0];
  const float* norm_w = (const float*)d_in[1];
  const float* norm_b = (const float*)d_in[2];
  const float* qkv_w  = (const float*)d_in[3];
  const float* qkv_b  = (const float*)d_in[4];
  const float* proj_w = (const float*)d_in[5];
  const float* proj_b = (const float*)d_in[6];
  float* out = (float*)d_out;

  const size_t act = (size_t)Bc * HW * C;  // == B*NH*HW*HD
  _Float16* hT   = (_Float16*)d_ws;
  _Float16* qT   = hT + act;
  _Float16* kT   = qT + act;
  _Float16* vbuf = kT + act;
  _Float16* aT   = vbuf + act;
  _Float16* wqh  = aT + act;
  _Float16* wph  = wqh + (size_t)3 * C * C;

  k_cvt_w<<<(3 * C * C + 255) / 256, 256, 0, stream>>>(qkv_w, proj_w, wqh, wph,
                                                       3 * C * C, C * C);
  k_gnorm<<<Bc * G, 256, 0, stream>>>(x, norm_w, norm_b, hT);
  k_qkv<<<dim3(HW / 128, (3 * C) / 128, Bc), 256, 0, stream>>>(wqh, qkv_b, hT,
                                                               qT, kT, vbuf);
  k_attn<<<(Bc * NH * (HW / 32)) / 8, 256, 0, stream>>>(qT, kT, vbuf, aT);
  k_proj<<<dim3(HW / 128, C / 128, Bc), 256, 0, stream>>>(wph, proj_b, aT, x, out);
}
